// MaxPatchGNN_6588479832607
// MI455X (gfx1250) — matrix operand order repaired and checked by reading, not compiled, not run on hardware
//
#include <hip/hip_runtime.h>
#include <hip/hip_bf16.h>

typedef __attribute__((ext_vector_type(16))) __bf16 v16bf;
typedef __attribute__((ext_vector_type(8)))  float  v8f;

#define HDIM 256
#define EPSBN 1e-5f

__device__ __forceinline__ unsigned short f2bf(float f) {
    unsigned int u = __float_as_uint(f);
    unsigned int r = u + 0x7FFFu + ((u >> 16) & 1u);   // round-to-nearest-even
    return (unsigned short)(r >> 16);
}

// ---------------- utility kernels ----------------

__global__ void k_zero_f32(float* __restrict__ p, long n) {
    long i = (long)blockIdx.x * blockDim.x + threadIdx.x;
    if (i < n) p[i] = 0.0f;
}

__global__ void k_deg_count(const int* __restrict__ dst, float* __restrict__ deg, int E) {
    int e = blockIdx.x * blockDim.x + threadIdx.x;
    if (e < E) atomicAdd(&deg[dst[e]], 1.0f);
}

__global__ void k_deg_recip(float* __restrict__ deg, int N) {
    int i = blockIdx.x * blockDim.x + threadIdx.x;
    if (i < N) deg[i] = 1.0f / fmaxf(deg[i], 1.0f);
}

// st[0..255] = scale, st[256..511] = shift:  y = z*s + t  ==  BN(z + bl)
__global__ void k_fold_bn(const float* __restrict__ bl, const float* __restrict__ g,
                          const float* __restrict__ b,  const float* __restrict__ rm,
                          const float* __restrict__ rv, float* __restrict__ st) {
    int n = threadIdx.x;
    if (n < HDIM) {
        float s = g[n] * rsqrtf(rv[n] + EPSBN);
        st[n]        = s;
        st[HDIM + n] = (bl[n] - rm[n]) * s + b[n];
    }
}

// wcat[2*din x 256] = [Wl ; Wr] in bf16
__global__ void k_pack_w(const float* __restrict__ Wl, const float* __restrict__ Wr,
                         unsigned short* __restrict__ wcat, int din) {
    int t = blockIdx.x * blockDim.x + threadIdx.x;
    int total = 2 * din * HDIM;
    if (t >= total) return;
    int r = t >> 8, n = t & 255;
    float v = (r < din) ? Wl[r * HDIM + n] : Wr[(r - din) * HDIM + n];
    wcat[t] = f2bf(v);
}

// scatter-add: agg[dst] += feat[src]  (float4 chunks; consecutive threads = consecutive chunks)
__global__ void k_aggregate(const float* __restrict__ feat, const int* __restrict__ srcI,
                            const int* __restrict__ dstI, float* __restrict__ agg,
                            int d, int E) {
    int t = blockIdx.x * blockDim.x + threadIdx.x;
    int chunks = d >> 2;
    int e = t / chunks;
    if (e >= E) return;
    int c4 = (t - e * chunks) << 2;
    const float4 v = *(const float4*)(feat + (long)srcI[e] * d + c4);
    float* ap = agg + (long)dstI[e] * d + c4;
    atomicAdd(ap + 0, v.x);
    atomicAdd(ap + 1, v.y);
    atomicAdd(ap + 2, v.z);
    atomicAdd(ap + 3, v.w);
}

// Xcat[node][0:din]=bf16(agg*recip), Xcat[node][din:2din]=bf16(feat)
__global__ void k_build_xcat(const float* __restrict__ agg, const float* __restrict__ recip,
                             const float* __restrict__ feat, unsigned short* __restrict__ xcat,
                             int din, long total) {
    long t = (long)blockIdx.x * blockDim.x + threadIdx.x;
    if (t >= total) return;
    int node = (int)(t / din);
    int f    = (int)(t - (long)node * din);
    long K   = 2L * din;
    xcat[(long)node * K + f]       = f2bf(agg[t] * recip[node]);
    xcat[(long)node * K + din + f] = f2bf(feat[t]);
}

// ---------------- WMMA GEMM + folded BN/ReLU epilogue ----------------
// grid.x = N/16 row tiles, grid.y = 2 column halves; 256 threads = 8 waves,
// wave w -> columns [blockIdx.y*128 + w*16, +16). A tile (16 x K bf16) staged in LDS.
__launch_bounds__(256)
__global__ void k_sage_wmma(const unsigned short* __restrict__ xcat,
                            const unsigned short* __restrict__ wcat,
                            const float* __restrict__ st,
                            float* __restrict__ out, int K, int do_relu) {
    __shared__ unsigned short ldsA[16 * 512];      // up to 16KB

    const int tid  = threadIdx.x;
    const int wave = tid >> 5;
    const int lane = tid & 31;
    const long mBase = (long)blockIdx.x * 16;
    const int  n0    = blockIdx.y * 128 + wave * 16;

    // cooperative, coalesced stage of A tile (b128 loads)
    {
        const uint4* sv = (const uint4*)(xcat + mBase * K);
        uint4* dv = (uint4*)ldsA;
        const int nvec = (16 * K) >> 3;
        for (int idx = tid; idx < nvec; idx += 256) dv[idx] = sv[idx];
    }
    __syncthreads();

    union AF { v16bf v; unsigned int u[8]; } a;
    union BF { v16bf v; uint4 q[2]; } bf;
    v8f c = {0.f, 0.f, 0.f, 0.f, 0.f, 0.f, 0.f, 0.f};

    const int M     = lane & 15;
    const int kbase = (lane < 16) ? 0 : 8;         // A-fragment K offset per half-wave

    for (int kk = 0; kk < K; kk += 32) {
        // A fragment: 16x32 bf16, ISA VGPR layout (pairs of consecutive K)
        #pragma unroll
        for (int v = 0; v < 8; ++v) {
            int klo = kk + kbase + ((v & 3) << 1) + ((v & 4) ? 16 : 0);
            a.u[v] = *(const unsigned int*)(ldsA + M * K + klo);
        }
        // B fragment: lane%16 = N column, 16 K values in same order as A
        const unsigned short* bcol = wcat + (long)kk * HDIM + n0 + (lane & 15);
        #pragma unroll
        for (int s = 0; s < 16; ++s) {
            int kb = kbase + (s & 7) + ((s & 8) ? 16 : 0);
            ((unsigned short*)&bf)[s] = bcol[(long)kb * HDIM];
        }

        c = __builtin_amdgcn_wmma_f32_16x16x32_bf16(
                false, a.v, false, bf.v, (short)0, c, false, false);
    }

    // epilogue: BN fold + optional ReLU.  C/D layout: lane%16 = N, VGPR r -> M = r + 8*(lane>=16)
    const int n = n0 + (lane & 15);
    const float s = st[n];
    const float t = st[HDIM + n];
    const int mOff = (lane < 16) ? 0 : 8;
    #pragma unroll
    for (int r = 0; r < 8; ++r) {
        float y = c[r] * s + t;
        if (do_relu) y = fmaxf(y, 0.0f);
        out[(mBase + r + mOff) * HDIM + n] = y;
    }
}

// ---------------- host-side launch ----------------

static inline size_t alignup(size_t x) { return (x + 255) & ~(size_t)255; }

extern "C" void kernel_launch(void* const* d_in, const int* in_sizes, int n_in,
                              void* d_out, int out_size, void* d_ws, size_t ws_size,
                              hipStream_t stream) {
    const int D0 = 128;
    const int N  = in_sizes[0] / D0;
    const int E  = in_sizes[1] / 2;

    const float* x   = (const float*)d_in[0];
    const int*   ei  = (const int*)d_in[1];
    const int*   src = ei;
    const int*   dst = ei + E;

    const float* Wl[3] = {(const float*)d_in[2],  (const float*)d_in[9],  (const float*)d_in[16]};
    const float* bl[3] = {(const float*)d_in[3],  (const float*)d_in[10], (const float*)d_in[17]};
    const float* Wr[3] = {(const float*)d_in[4],  (const float*)d_in[11], (const float*)d_in[18]};
    const float* g[3]  = {(const float*)d_in[5],  (const float*)d_in[12], (const float*)d_in[19]};
    const float* b[3]  = {(const float*)d_in[6],  (const float*)d_in[13], (const float*)d_in[20]};
    const float* rm[3] = {(const float*)d_in[7],  (const float*)d_in[14], (const float*)d_in[21]};
    const float* rv[3] = {(const float*)d_in[8],  (const float*)d_in[15], (const float*)d_in[22]};

    // workspace carve-out (~155 MB); agg/h/xcat all L2-resident on MI455X (192MB L2)
    char* p = (char*)d_ws;
    float* deg = (float*)p;                 p += alignup((size_t)N * 4);
    float* agg = (float*)p;                 p += alignup((size_t)N * HDIM * 4);
    float* h   = (float*)p;                 p += alignup((size_t)N * HDIM * 4);
    unsigned short* xcat = (unsigned short*)p; p += alignup((size_t)N * 512 * 2);
    unsigned short* wc[3];
    wc[0] = (unsigned short*)p;             p += alignup((size_t)256 * HDIM * 2);
    wc[1] = (unsigned short*)p;             p += alignup((size_t)512 * HDIM * 2);
    wc[2] = (unsigned short*)p;             p += alignup((size_t)512 * HDIM * 2);
    float* st[3];
    st[0] = (float*)p;                      p += alignup(512 * 4);
    st[1] = (float*)p;                      p += alignup(512 * 4);
    st[2] = (float*)p;                      p += alignup(512 * 4);

    const int BT = 256;

    // degree (edge-structure dependent only, but recompute each call for determinism)
    k_zero_f32<<<(N + BT - 1) / BT, BT, 0, stream>>>(deg, N);
    k_deg_count<<<(E + BT - 1) / BT, BT, 0, stream>>>(dst, deg, E);
    k_deg_recip<<<(N + BT - 1) / BT, BT, 0, stream>>>(deg, N);

    // fold BN + pack weights to bf16
    for (int l = 0; l < 3; ++l) {
        int din = (l == 0) ? D0 : HDIM;
        k_fold_bn<<<1, 256, 0, stream>>>(bl[l], g[l], b[l], rm[l], rv[l], st[l]);
        int tw = 2 * din * HDIM;
        k_pack_w<<<(tw + BT - 1) / BT, BT, 0, stream>>>(Wl[l], Wr[l], wc[l], din);
    }

    const int mTiles = N / 16;   // N = 50000 -> 3125 exact

    for (int l = 0; l < 3; ++l) {
        int din = (l == 0) ? D0 : HDIM;
        int K   = 2 * din;
        const float* feat = (l == 0) ? x : h;
        float* outp = (l == 2) ? (float*)d_out : h;

        long na = (long)N * din;
        k_zero_f32<<<(unsigned)((na + BT - 1) / BT), BT, 0, stream>>>(agg, na);

        long tedge = (long)E * (din >> 2);
        k_aggregate<<<(unsigned)((tedge + BT - 1) / BT), BT, 0, stream>>>(
            feat, src, dst, agg, din, E);

        k_build_xcat<<<(unsigned)((na + BT - 1) / BT), BT, 0, stream>>>(
            agg, deg, feat, xcat, din, na);

        dim3 grid(mTiles, 2);
        k_sage_wmma<<<grid, 256, 0, stream>>>(xcat, wc[l], st[l], outp, K, (l < 2) ? 1 : 0);
    }
}
